// ScaleDotProductAttention_43542378447571
// MI455X (gfx1250) — matrix operand order repaired and checked by reading, not compiled, not run on hardware
//
#include <hip/hip_runtime.h>

// ---------------------------------------------------------------------------
// Attention fwd for MI455X (gfx1250, wave32):
//   scores = QK^T * 0.125 ; masked -> 1e-9 ; attn = softmax(scores) ; ctx = attn @ V
// Cooperative 4-wave block: K/V tiles staged in LDS once per block (coalesced
// b128), fragments built from LDS. Two-pass flash-style softmax (per-lane
// online stats, one merge tree). attn written b128 via LDS transpose.
// All GEMMs: v_wmma_f32_16x16x32_bf16 (f32 accumulate).
// ---------------------------------------------------------------------------

typedef __attribute__((ext_vector_type(16))) __bf16 v16bf;
typedef __attribute__((ext_vector_type(8)))  float  v8f;

#define kB 8
#define kH 16
#define kS 1024
#define kD 64
#define kScale 0.125f    /* 1/sqrt(64) */
#define kPstride 36      /* P staging row stride (floats): 16B-aligned */
#define kTstride 68      /* K/V tile row stride (floats): 16B-aligned, bank-spread */

static __device__ __forceinline__ __bf16 f2bf(float f) {
  return (__bf16)f;      // native v_cvt_*bf16_f32 (RNE) on gfx1250
}

// A-fragment (16x32 bf16) from a row-major f32 row pointer (&M[row][kk]).
// Layout: lanes 0-15 hold K 0..15, lanes 16-31 hold K 16..31 (matches load_fragB).
static __device__ __forceinline__ v16bf load_fragA(const float* row_ptr, int half) {
  v16bf v;
#pragma unroll
  for (int jg = 0; jg < 2; ++jg) {
    const float4* p = reinterpret_cast<const float4*>(row_ptr + half * 16 + jg * 8);
    float4 x0 = p[0], x1 = p[1];
    v[jg*8+0]=f2bf(x0.x); v[jg*8+1]=f2bf(x0.y); v[jg*8+2]=f2bf(x0.z); v[jg*8+3]=f2bf(x0.w);
    v[jg*8+4]=f2bf(x1.x); v[jg*8+5]=f2bf(x1.y); v[jg*8+6]=f2bf(x1.z); v[jg*8+7]=f2bf(x1.w);
  }
  return v;
}

// B-fragment (32x16 bf16), column n = lane&15 from a row-major row pointer.
// Layout (this code's order, as load_fragA): lanes 0-15 hold K 0..15, lanes 16-31 hold K 16..31 (2 per VGPR).
static __device__ __forceinline__ v16bf load_fragB(const float* row_ptr, int half) {
  v16bf v;
  const float4* p = reinterpret_cast<const float4*>(row_ptr + half * 16);
#pragma unroll
  for (int q = 0; q < 4; ++q) {
    float4 x = p[q];
    v[q*4+0]=f2bf(x.x); v[q*4+1]=f2bf(x.y); v[q*4+2]=f2bf(x.z); v[q*4+3]=f2bf(x.w);
  }
  return v;
}

// Cooperative stage of a 32x64 f32 tile global -> LDS (512 float4, 4/thread).
static __device__ __forceinline__ void stage_tile32(float* __restrict__ dst,
                                                    const float* __restrict__ src,
                                                    int tid) {
#pragma unroll
  for (int t = 0; t < 4; ++t) {
    const int idx = tid + t * 128;          // 0..511
    const int row = idx >> 4;               // 16 float4 per 64-float row
    const int c4  = idx & 15;
    const float4 x = reinterpret_cast<const float4*>(src + (size_t)row * kD)[c4];
    *reinterpret_cast<float4*>(dst + row * kTstride + c4 * 4) = x;
  }
}

// One 16x16 score tile vs K rows staged in LDS; scaled + masked.
// C/D layout: s[r] holds row m = r + 8*half, col n = lane&15.
static __device__ __forceinline__ void score_tile(
    const v16bf& aq0, const v16bf& aq1,
    const float* Krow,              /* LDS: &Kt[(h2*16+ln)*kTstride] */
    const unsigned char* maskElem,  /* global: mask_b + q0*kS + k0 + ln */
    int half, float s[8]) {
  v8f c = {};
  v16bf b0 = load_fragB(Krow + 0, half);
  c = __builtin_amdgcn_wmma_f32_16x16x32_bf16(false, aq0, false, b0, (short)0, c, false, false);
  v16bf b1 = load_fragB(Krow + 32, half);
  c = __builtin_amdgcn_wmma_f32_16x16x32_bf16(false, aq1, false, b1, (short)0, c, false, false);
#pragma unroll
  for (int r = 0; r < 8; ++r) {
    const int m = r + 8 * half;
    float sc = c[r] * kScale;
    // Faithful to reference: masked scores become 1e-9 (NOT -inf).
    s[r] = maskElem[(size_t)m * kS] ? 1e-9f : sc;
  }
}

__global__ __launch_bounds__(128)
void ScaleDotProductAttention_43542378447571_kernel(
    const float* __restrict__ Q, const float* __restrict__ K,
    const float* __restrict__ V, const unsigned char* __restrict__ mask,
    float* __restrict__ ctx_out, float* __restrict__ attn_out) {
  __shared__ float Kt_s[32 * kTstride];      // shared K tile (32 keys x 64)
  __shared__ float Vt_s[32 * kTstride];      // shared V tile (32 keys x 64)
  __shared__ float Pst[4][16 * kPstride];    // per-wave 16x32 P staging

  const int tid  = threadIdx.x;
  const int lane = tid & 31;
  const int wave = tid >> 5;
  const int ln   = lane & 15;
  const int half = lane >> 4;

  const int nQBlk = kS / 64;                 // 16 blocks of 64 queries per (b,h)
  const int bh    = blockIdx.x / nQBlk;      // 0..127
  const int qblk  = blockIdx.x % nQBlk;
  const int b     = bh / kH;
  const int q0    = qblk * 64 + wave * 16;   // this wave's 16-query strip

  const float* Qbh = Q + (size_t)bh * kS * kD;
  const float* Kbh = K + (size_t)bh * kS * kD;
  const float* Vbh = V + (size_t)bh * kS * kD;
  const unsigned char* maskQ = mask + (size_t)b * kS * kS + (size_t)q0 * kS;
  float* attnQ = attn_out + ((size_t)bh * kS + q0) * kS;
  float* ctxQ  = ctx_out  + ((size_t)bh * kS + q0) * kD;

  // Stage Q tile (16x64) once as two bf16 A fragments (K chunks 0-31, 32-63).
  const float* qrow = Qbh + (size_t)(q0 + ln) * kD;
  v16bf aq0 = load_fragA(qrow +  0, half);
  v16bf aq1 = load_fragA(qrow + 32, half);

  // ---- Pass 1: PER-LANE online softmax stats (no cross-lane ops in loop) ----
  float lm[8], ls[8];
#pragma unroll
  for (int r = 0; r < 8; ++r) { lm[r] = -3.0e38f; ls[r] = 0.0f; }

  for (int kt = 0; kt < kS; kt += 32) {
    stage_tile32(Kt_s, Kbh + (size_t)kt * kD, tid);
    if (kt + 32 < kS)   // pull next 8KB K tile toward L0 (global_prefetch_b8)
      __builtin_prefetch(Kbh + (size_t)(kt + 32) * kD + tid * 16, 0, 3);
    __syncthreads();
#pragma unroll
    for (int h2 = 0; h2 < 2; ++h2) {
      float s[8];
      score_tile(aq0, aq1, Kt_s + (size_t)(h2 * 16 + ln) * kTstride,
                 maskQ + kt + h2 * 16 + ln, half, s);
#pragma unroll
      for (int r = 0; r < 8; ++r) {
        const float nm = fmaxf(lm[r], s[r]);
        ls[r] = ls[r] * __expf(lm[r] - nm) + __expf(s[r] - nm);
        lm[r] = nm;
      }
    }
    __syncthreads();
  }

  // Single (max,sum) merge tree across the 16-lane row group.
#pragma unroll
  for (int r = 0; r < 8; ++r) {
#pragma unroll
    for (int d = 1; d < 16; d <<= 1) {
      const float om = __shfl_xor(lm[r], d, 16);
      const float os = __shfl_xor(ls[r], d, 16);
      const float nm = fmaxf(lm[r], om);
      ls[r] = ls[r] * __expf(lm[r] - nm) + os * __expf(om - nm);
      lm[r] = nm;
    }
  }

  float inv[8];
#pragma unroll
  for (int r = 0; r < 8; ++r) inv[r] = 1.0f / ls[r];

  // ---- Pass 2: recompute scores, emit attn (b128 via LDS), ctx += P @ V ----
  v8f acc[4] = {};                    // 4 d-column tiles of 16, f32 accum
  float* myP = &Pst[wave][0];

  for (int kt = 0; kt < kS; kt += 32) {
    stage_tile32(Kt_s, Kbh + (size_t)kt * kD, tid);
    stage_tile32(Vt_s, Vbh + (size_t)kt * kD, tid);
    if (kt + 32 < kS) {
      __builtin_prefetch(Kbh + (size_t)(kt + 32) * kD + tid * 16, 0, 3);
      __builtin_prefetch(Vbh + (size_t)(kt + 32) * kD + tid * 16, 0, 3);
    }
    __syncthreads();

#pragma unroll
    for (int h2 = 0; h2 < 2; ++h2) {
      float s[8];
      score_tile(aq0, aq1, Kt_s + (size_t)(h2 * 16 + ln) * kTstride,
                 maskQ + kt + h2 * 16 + ln, half, s);
#pragma unroll
      for (int r = 0; r < 8; ++r) {
        const int m = r + 8 * half;
        myP[m * kPstride + h2 * 16 + ln] = __expf(s[r] - lm[r]) * inv[r];
      }
    }
    // C-layout -> row-major transpose goes through LDS: fence DS stores.
    asm volatile("s_wait_dscnt 0" ::: "memory");

    // attn writeback: lane = row ln, cols half*16..+15 -> 4x global_store_b128.
    {
      const float4* lrow = reinterpret_cast<const float4*>(myP + ln * kPstride + half * 16);
      float4* grow = reinterpret_cast<float4*>(attnQ + (size_t)ln * kS + kt + half * 16);
#pragma unroll
      for (int q = 0; q < 4; ++q) grow[q] = lrow[q];
    }

    // P tile (16 rows x 32 keys) as bf16 A fragment: 4x ds_load_b128.
    v16bf ap;
    {
      const float4* p0 = reinterpret_cast<const float4*>(myP + ln * kPstride + half * 16);
      const float4* p1 = reinterpret_cast<const float4*>(myP + ln * kPstride + half * 16 + 8);
      float4 x0 = p0[0], x1 = p0[1], y0 = p1[0], y1 = p1[1];
      ap[0]=f2bf(x0.x);  ap[1]=f2bf(x0.y);  ap[2]=f2bf(x0.z);  ap[3]=f2bf(x0.w);
      ap[4]=f2bf(x1.x);  ap[5]=f2bf(x1.y);  ap[6]=f2bf(x1.z);  ap[7]=f2bf(x1.w);
      ap[8]=f2bf(y0.x);  ap[9]=f2bf(y0.y);  ap[10]=f2bf(y0.z); ap[11]=f2bf(y0.w);
      ap[12]=f2bf(y1.x); ap[13]=f2bf(y1.y); ap[14]=f2bf(y1.z); ap[15]=f2bf(y1.w);
    }

    // V B-fragments read column-strided from the shared LDS tile.
#pragma unroll
    for (int dt = 0; dt < 4; ++dt) {
      v16bf bv;
#pragma unroll
      for (int j = 0; j < 16; ++j) {
        const int k = half * 16 + j;                // B layout K index
        bv[j] = f2bf(Vt_s[k * kTstride + dt * 16 + ln]);
      }
      acc[dt] = __builtin_amdgcn_wmma_f32_16x16x32_bf16(false, ap, false, bv,
                                                        (short)0, acc[dt], false, false);
    }
    __syncthreads();
  }

  // ---- Write context (f32), C/D layout: row r+8*half, col dt*16 + ln ----
#pragma unroll
  for (int dt = 0; dt < 4; ++dt) {
#pragma unroll
    for (int r = 0; r < 8; ++r) {
      ctxQ[(size_t)(r + 8 * half) * kD + dt * 16 + ln] = acc[dt][r];
    }
  }
}

extern "C" void kernel_launch(void* const* d_in, const int* in_sizes, int n_in,
                              void* d_out, int out_size, void* d_ws, size_t ws_size,
                              hipStream_t stream) {
  (void)in_sizes; (void)n_in; (void)out_size; (void)d_ws; (void)ws_size;
  const float* Q = (const float*)d_in[0];
  const float* K = (const float*)d_in[1];
  const float* V = (const float*)d_in[2];
  // jax bool array -> 1 byte per element on device
  const unsigned char* mask = (const unsigned char*)d_in[3];

  float* ctx  = (float*)d_out;                                   // [B,H,S,D]
  float* attn = (float*)d_out + (size_t)kB * kH * kS * kD;       // [B,H,S,S]

  const int grid = kB * kH * (kS / 64);   // 2048 blocks, 4 waves each
  ScaleDotProductAttention_43542378447571_kernel<<<grid, 128, 0, stream>>>(
      Q, K, V, mask, ctx, attn);
}
